// SimpleHarmonicRNN_89756226552394
// MI455X (gfx1250) — matrix operand order repaired and checked by reading, not compiled, not run on hardware
//
#include <hip/hip_runtime.h>

typedef _Float16 half8 __attribute__((ext_vector_type(8)));
typedef _Float16 v16h  __attribute__((ext_vector_type(16)));
typedef float    v8f   __attribute__((ext_vector_type(8)));
typedef int      v4i   __attribute__((ext_vector_type(4)));
typedef v4i __attribute__((address_space(3))) as3_v4i;

#define T_STEPS 1024
#define Bsz     256
#define INsz    256
#define Hsz     512
#define OUTsz   20
#define KTOT    768           // IN + H
#define APITCH  776           // halves per LDS row (768 + 8 pad -> conflict-free b128)
#define BC      16            // batch rows per workgroup
#define NTHREADS 512          // 16 waves of 32
#define DT_     0.01f
#define THETA_  1.0f
#define GAMMA_  0.9f

// d_out layout (floats): outs[1,B,OUT], zf[B,H], uf[B,H], ouf[B,OUT], spk
#define OUTS_OFF 0
#define ZF_OFF   5120
#define UF_OFF   136192
#define OUF_OFF  267264
#define SPK_OFF  272384

// d_ws layout (halves): WT_hid[512][768] (N-major), WT_out[32][512] (N-major, rows>=20 zero)
#define WTO_OFF_H (512*768)

// ---- optional CDNA5 async global->LDS path (guarded so either way compiles) ------
#if defined(__has_builtin)
#  if __has_builtin(__builtin_amdgcn_global_load_async_to_lds_b128)
#    define HAVE_ASYNC_LDS 1
#  endif
#endif
#ifndef HAVE_ASYNC_LDS
#  define HAVE_ASYNC_LDS 0
#endif

#if defined(__has_builtin) && __has_builtin(__builtin_amdgcn_s_wait_asynccnt)
#  define WAIT_ASYNC0() __builtin_amdgcn_s_wait_asynccnt(0)
#else
#  define WAIT_ASYNC0() asm volatile("s_wait_asynccnt 0x0" ::: "memory")
#endif

#if HAVE_ASYNC_LDS
#define ASYNC_LOAD_B128(gp, lp)                                              \
    __builtin_amdgcn_global_load_async_to_lds_b128(                          \
        (v4i*)(gp), (as3_v4i*)(lp), 0, 0)
#endif

// Defeat LICM without destroying address-space provenance: an opaque zero offset.
// (Laundering the pointer itself turns global_load into flat_load.)
#define LAUNDER_I32(v) asm volatile("" : "+v"(v))

// ---------------- Prologue: convert + transpose weights to f16, zero spike slot ----
__global__ void k_prep(const float* __restrict__ Wh, const float* __restrict__ Wo,
                       _Float16* __restrict__ ws, float* __restrict__ dout) {
    int idx = blockIdx.x * 256 + threadIdx.x;          // grid covers 512*768 + 32*512
    if (idx < 512 * 768) {
        int n = idx / 768, k = idx % 768;              // WT[n][k] = W_hid[k][n]
        ws[idx] = (_Float16)Wh[k * 512 + n];
    }
    int j = idx - 512 * 768;
    if (j >= 0 && j < 32 * 512) {
        int n = j / 512, k = j % 512;                  // WTo[n][k] = W_out[k][n], pad n>=20
        ws[WTO_OFF_H + j] = (n < OUTsz) ? (_Float16)Wo[k * OUTsz + n] : (_Float16)0.0f;
    }
    if (idx == 0) dout[SPK_OFF] = 0.0f;
}

// ---------------- Main persistent RNN kernel: 16 WGs x 512 threads (16 waves) ------
__global__ void __launch_bounds__(NTHREADS, 1)
k_rnn(const float* __restrict__ x, const float* __restrict__ omega,
      const float* __restrict__ boff, const float* __restrict__ tau,
      const _Float16* __restrict__ ws, float* __restrict__ dout) {
    __shared__ _Float16 Ash[BC * APITCH];   // [16][776]: cols 0..255 = x_t(f16), 256..767 = z(f16)
#if HAVE_ASYNC_LDS
    __shared__ float Xst[2][BC * INsz];     // double-buffered raw x tile (async-filled)
#endif
    __shared__ float red[16];

    const int tid   = threadIdx.x;
    const int w     = tid >> 5;             // wave id 0..15
    const int lane  = tid & 31;
    const int lhalf = lane >> 4;            // 0/1 (lane half, per WMMA layouts)
    const int l15   = lane & 15;
    const int b0    = blockIdx.x * BC;

    const _Float16* __restrict__ WT  = ws;
    const _Float16* __restrict__ WTo = ws + WTO_OFF_H;

    // Per-lane column params for the 2 N-tiles this wave owns (cols w*32 + t*16 + l15)
    int   ncol[2];
    float om2v[2], bbv[2];
#pragma unroll
    for (int t = 0; t < 2; ++t) {
        int n = w * 32 + t * 16 + l15;
        ncol[t] = n;
        float om = fabsf(omega[n]);
        om2v[t] = om * om;
        bbv[t]  = fabsf(boff[n]);
    }
    float alphav[2] = {0.0f, 0.0f};
    if (w == 0) {
#pragma unroll
        for (int t2 = 0; t2 < 2; ++t2) {
            int o = t2 * 16 + l15;
            alphav[t2] = (o < OUTsz) ? expf(-DT_ / fabsf(tau[o])) : 0.0f;
        }
    }

    // Oscillator state in WMMA C-fragment layout: row m = r + 8*lhalf, col = ncol[t]
    float u_s[2][8], v_s[2][8], a_s[2][8];
#pragma unroll
    for (int t = 0; t < 2; ++t)
#pragma unroll
        for (int r = 0; r < 8; ++r) { u_s[t][r] = 0.f; v_s[t][r] = 0.f; a_s[t][r] = 0.f; }
    float ou_s[2][8];
#pragma unroll
    for (int t2 = 0; t2 < 2; ++t2)
#pragma unroll
        for (int r = 0; r < 8; ++r) ou_s[t2][r] = 0.f;

    // Zero the z region of LDS (cols 256..767)
    {
        half8 hz;
#pragma unroll
        for (int i = 0; i < 8; ++i) hz[i] = (_Float16)0.0f;
        for (int c = tid; c < (BC * Hsz) / 8; c += NTHREADS) {
            int m = c >> 6, off = (c & 63) * 8;
            *(half8*)&Ash[m * APITCH + INsz + off] = hz;
        }
    }

#if HAVE_ASYNC_LDS
    // Kick off async staging of x_0 into buffer 0 (each thread owns 32B it will consume)
    {
        const float* gsrc = x + (size_t)b0 * INsz + (size_t)tid * 8;
        float*       ldst = &Xst[0][tid * 8];
        ASYNC_LOAD_B128(gsrc,     ldst);
        ASYNC_LOAD_B128(gsrc + 4, ldst + 4);
    }
#endif

    // Fragment base pointers (A: lanes 0-15 = rows 0-15 K+0..7/16..23; lanes 16-31 K+8..15/24..31)
    const _Float16* arow = &Ash[l15 * APITCH + lhalf * 8];

    float spkacc = 0.0f;

    for (int tt = 0; tt < T_STEPS; ++tt) {
        // -------- Phase A: stage x_t (16x256 f32 -> f16) into LDS --------------
        {
            int m  = tid >> 5;
            int cb = (tid & 31) * 8;
#if HAVE_ASYNC_LDS
            WAIT_ASYNC0();
            const float* sp = &Xst[tt & 1][tid * 8];
            float4 f0 = *(const float4*)(sp + 0);
            float4 f1 = *(const float4*)(sp + 4);
#else
            const float* xp = x + ((size_t)tt * Bsz + b0 + m) * INsz + cb;
            float4 f0 = *(const float4*)(xp + 0);
            float4 f1 = *(const float4*)(xp + 4);
            __builtin_prefetch(xp + (size_t)Bsz * INsz, 0, 1);   // next step's x
#endif
            half8 h0;
            h0[0]=(_Float16)f0.x; h0[1]=(_Float16)f0.y; h0[2]=(_Float16)f0.z; h0[3]=(_Float16)f0.w;
            h0[4]=(_Float16)f1.x; h0[5]=(_Float16)f1.y; h0[6]=(_Float16)f1.z; h0[7]=(_Float16)f1.w;
            *(half8*)&Ash[m * APITCH + cb] = h0;
        }
#if HAVE_ASYNC_LDS
        if (tt + 1 < T_STEPS) {  // issue async staging of x_{t+1} into the other buffer
            const float* gsrc = x + ((size_t)(tt + 1) * Bsz + b0) * INsz + (size_t)tid * 8;
            float*       ldst = &Xst[(tt + 1) & 1][tid * 8];
            ASYNC_LOAD_B128(gsrc,     ldst);
            ASYNC_LOAD_B128(gsrc + 4, ldst + 4);
        }
#endif
        __syncthreads();

        // -------- Phase B: inp = [x_t|z] @ W_hid  (per wave: 16x32 slice, K=768) ----
        // Opaque zero offset: keeps W loads as per-step global_load_b128 from L2
        // (otherwise LICM hoists 1.5KB/lane of W out of the time loop and spills).
        int woff = 0;
        LAUNDER_I32(woff);
        const _Float16* wrow[2] = { WT + ncol[0] * KTOT + lhalf * 8 + woff,
                                    WT + ncol[1] * KTOT + lhalf * 8 + woff };

        v8f acc[2];
#pragma unroll
        for (int t = 0; t < 2; ++t)
#pragma unroll
            for (int i = 0; i < 8; ++i) acc[t][i] = 0.0f;

        for (int kk = 0; kk < KTOT; kk += 32) {
            half8 alo = *(const half8*)(arow + kk);
            half8 ahi = *(const half8*)(arow + kk + 16);
            v16h af = __builtin_shufflevector(alo, ahi, 0,1,2,3,4,5,6,7,8,9,10,11,12,13,14,15);
#pragma unroll
            for (int t = 0; t < 2; ++t) {
                half8 blo = *(const half8*)(wrow[t] + kk);
                half8 bhi = *(const half8*)(wrow[t] + kk + 16);
                v16h bf = __builtin_shufflevector(blo, bhi, 0,1,2,3,4,5,6,7,8,9,10,11,12,13,14,15);
                acc[t] = __builtin_amdgcn_wmma_f32_16x16x32_f16(
                    false, af, false, bf, (short)0, acc[t], false, false);
            }
        }
        __syncthreads();   // all waves done reading z(t-1) before overwriting it

        // -------- Phase C: oscillator + spike, write z_new to LDS ---------------
#pragma unroll
        for (int t = 0; t < 2; ++t) {
            float om2 = om2v[t], bb = bbv[t];
#pragma unroll
            for (int r = 0; r < 8; ++r) {
                float inp = acc[t][r];
                float av  = a_s[t][r];
                float bc  = bb + av;
                float un  = u_s[t][r] + DT_ * v_s[t][r];
                float vn  = v_s[t][r] + DT_ * (inp - 2.0f * bc * v_s[t][r] - om2 * un);
                float zn  = (un - (THETA_ + av)) > 0.0f ? 1.0f : 0.0f;
                a_s[t][r] = GAMMA_ * av + zn;
                u_s[t][r] = un;
                v_s[t][r] = vn;
                spkacc   += zn;
                Ash[(r + 8 * lhalf) * APITCH + INsz + ncol[t]] = (_Float16)zn;
            }
        }
        __syncthreads();   // z_new complete

        // -------- Phase D: wave 0 readout GEMM z_new @ W_out, leaky filter ------
        if (w == 0) {
            v8f oacc[2];
#pragma unroll
            for (int t2 = 0; t2 < 2; ++t2)
#pragma unroll
                for (int i = 0; i < 8; ++i) oacc[t2][i] = 0.0f;
            const _Float16* zrow = &Ash[l15 * APITCH + INsz + lhalf * 8];
            int woff2 = 0;
            LAUNDER_I32(woff2);
            const _Float16* wro = WTo + l15 * Hsz + lhalf * 8 + woff2;  // + t2*16*Hsz + kk
            for (int kk = 0; kk < Hsz; kk += 32) {
                half8 alo = *(const half8*)(zrow + kk);
                half8 ahi = *(const half8*)(zrow + kk + 16);
                v16h af = __builtin_shufflevector(alo, ahi, 0,1,2,3,4,5,6,7,8,9,10,11,12,13,14,15);
#pragma unroll
                for (int t2 = 0; t2 < 2; ++t2) {
                    const _Float16* wrp = wro + t2 * 16 * Hsz + kk;
                    half8 blo = *(const half8*)(wrp);
                    half8 bhi = *(const half8*)(wrp + 16);
                    v16h bf = __builtin_shufflevector(blo, bhi, 0,1,2,3,4,5,6,7,8,9,10,11,12,13,14,15);
                    oacc[t2] = __builtin_amdgcn_wmma_f32_16x16x32_f16(
                        false, af, false, bf, (short)0, oacc[t2], false, false);
                }
            }
#pragma unroll
            for (int t2 = 0; t2 < 2; ++t2) {
                float al = alphav[t2];
#pragma unroll
                for (int r = 0; r < 8; ++r)
                    ou_s[t2][r] = al * ou_s[t2][r] + (1.0f - al) * oacc[t2][r];
            }
        }
        // no barrier needed here: next write to the z region is after two barriers
    }

    __syncthreads();

    // -------- Epilogue: write zf, uf, ouf/outs, spike total ---------------------
    for (int e = tid; e < BC * Hsz; e += NTHREADS) {
        int m = e >> 9, n = e & 511;
        dout[ZF_OFF + (size_t)(b0 + m) * Hsz + n] = (float)Ash[m * APITCH + INsz + n];
    }
#pragma unroll
    for (int t = 0; t < 2; ++t)
#pragma unroll
        for (int r = 0; r < 8; ++r) {
            int m = r + 8 * lhalf;
            dout[UF_OFF + (size_t)(b0 + m) * Hsz + ncol[t]] = u_s[t][r];
        }
    if (w == 0) {
#pragma unroll
        for (int t2 = 0; t2 < 2; ++t2) {
            int o = t2 * 16 + l15;
            if (o < OUTsz) {
#pragma unroll
                for (int r = 0; r < 8; ++r) {
                    int m = r + 8 * lhalf;
                    dout[OUF_OFF  + (b0 + m) * OUTsz + o] = ou_s[t2][r];
                    dout[OUTS_OFF + (b0 + m) * OUTsz + o] = ou_s[t2][r];
                }
            }
        }
    }
#pragma unroll
    for (int off = 16; off > 0; off >>= 1) spkacc += __shfl_down(spkacc, off);
    if (lane == 0) red[w] = spkacc;
    __syncthreads();
    if (tid == 0) {
        float s = 0.0f;
#pragma unroll
        for (int i = 0; i < 16; ++i) s += red[i];
        atomicAdd(&dout[SPK_OFF], s);
    }
}

extern "C" void kernel_launch(void* const* d_in, const int* in_sizes, int n_in,
                              void* d_out, int out_size, void* d_ws, size_t ws_size,
                              hipStream_t stream) {
    const float* x     = (const float*)d_in[0];
    const float* Wh    = (const float*)d_in[1];
    const float* omega = (const float*)d_in[2];
    const float* boff  = (const float*)d_in[3];
    const float* Wo    = (const float*)d_in[4];
    const float* tau   = (const float*)d_in[5];
    float*     dout = (float*)d_out;
    _Float16*  wsh  = (_Float16*)d_ws;

    // 512*768 + 32*512 = 409600 elements -> 1600 blocks of 256
    k_prep<<<1600, 256, 0, stream>>>(Wh, Wo, wsh, dout);
    // 16 independent persistent WGs (16 batch rows each), 16 waves of 32
    k_rnn<<<16, NTHREADS, 0, stream>>>(x, omega, boff, tau, wsh, dout);
}
